// CustomMultiHeadSelfAttention_8443905704242
// MI455X (gfx1250) — matrix operand order repaired and checked by reading, not compiled, not run on hardware
//
#include <hip/hip_runtime.h>

// ---------------------------------------------------------------------------
// MI455X (gfx1250) multi-head self-attention forward.
//  - fp32 -> bf16 (RNE) conversion, all matmuls on v_wmma_f32_16x16x32_bf16
//    with fp32 accumulation.
//  - GEMMs: 32x64 tile per wave, software-pipelined (double-buffered frags).
//  - Attention: flash-style online softmax; K/V chunks staged into LDS by the
//    Tensor Data Mover (tensor_load_to_lds + s_wait_tensorcnt), double
//    buffered, shared by 8 waves per block.
// ---------------------------------------------------------------------------

typedef __attribute__((ext_vector_type(16))) __bf16 v16bf;
typedef __attribute__((ext_vector_type(4)))  __bf16 v4bf;
typedef __attribute__((ext_vector_type(8)))  float  v8f;
typedef __attribute__((ext_vector_type(4)))  unsigned v4u;
typedef __attribute__((ext_vector_type(8)))  int    v8i;
typedef __attribute__((ext_vector_type(4)))  int    v4i;

#define BATCH 8
#define SEQ   1024
#define EMB   1024
#define EMB3  3072
#define NHEAD 16
#define HDIM  64

// ---------------- fragment loaders (contiguous k = 16*(lane/16) + s, A and B alike) ----------

// A-matrix 16x32 bf16 tile from row-major memory, leading dim ld (elements).
__device__ inline v16bf load_a16(const __bf16* __restrict__ p0, int ld) {
  int lane = threadIdx.x & 31;
  int m    = lane & 15;
  int kb   = (lane >> 4) << 4;           // 0 or 16
  const __bf16* p = p0 + (size_t)m * ld + kb;
  union { v16bf v; unsigned u[8]; } f;
#pragma unroll
  for (int i = 0; i < 8; ++i)
    f.u[i] = *(const unsigned*)(p + 2 * i);          // K = kb + 0..15
  return f.v;
}

// B-matrix 32x16 bf16 tile, element (k,n) = p0[n*ld + k] (k contiguous).
__device__ inline v16bf load_b_rm(const __bf16* __restrict__ p0, int ld) {
  int lane = threadIdx.x & 31;
  int n    = lane & 15;
  int kb   = (lane >> 4) << 4;           // 0 or 16
  const __bf16* p = p0 + (size_t)n * ld + kb;
  union { v16bf v; unsigned u[8]; } f;
#pragma unroll
  for (int i = 0; i < 8; ++i)
    f.u[i] = *(const unsigned*)(p + 2 * i);
  return f.v;
}

// B-matrix 32x16 bf16 tile, element (k,n) = p0[k*ld + n] (k strided).
__device__ inline v16bf load_b_cm(const __bf16* __restrict__ p0, int ld) {
  int lane = threadIdx.x & 31;
  int n    = lane & 15;
  int kb   = (lane >> 4) << 4;           // 0 or 16
  union { v16bf v; __bf16 e[16]; } f;
#pragma unroll
  for (int i = 0; i < 16; ++i)
    f.e[i] = p0[(size_t)(kb + i) * ld + n];
  return f.v;
}

// ---------------- TDM: 2-D bf16 tile (tile_d0 x tile_d1 rows) -> LDS -------
// D# built per CDNA5 ISA ch.8: group0 {count=1, lds_addr, global_addr,
// type=2}; group1 {data_size=2B, dims, tile dims, row stride}; trailing
// groups zero (tile_dim3/4 = 0 => 2-D tensor). This toolchain declares the
// 6-arg builtin (g0 v4u, g1 v8i, g2 v4i, g3 v4i, g4 v8i, cpol).

__device__ inline void tdm_load_2d(unsigned lds_off, const void* gptr,
                                   unsigned tile_d0, unsigned tile_d1,
                                   unsigned row_stride_elems) {
  unsigned long long ga = (unsigned long long)(size_t)gptr;
  v4u g0;
  g0[0] = 1u;                                             // count=1, user desc
  g0[1] = lds_off;                                        // lds_addr (bytes)
  g0[2] = (unsigned)ga;                                   // global_addr[31:0]
  g0[3] = (unsigned)((ga >> 32) & 0x01FFFFFFu) | (2u << 30); // addr[56:32]|type=2
  v8i g1;
  g1[0] = (1 << 16);                                      // data_size = 2 bytes
  g1[1] = (int)(tile_d0 << 16);                           // tensor_dim0 = tile_d0
  g1[2] = (int)(tile_d1 << 16);                           // tensor_dim1 = tile_d1
  g1[3] = (int)(tile_d0 << 16);                           // tile_dim0
  g1[4] = (int)tile_d1;                                   // tile_dim1 (tile_dim2=0)
  g1[5] = (int)row_stride_elems;                          // tensor_dim0_stride lo
  g1[6] = 0;                                              // stride hi / dim1_stride lo
  g1[7] = 0;
  v4i z4 = {0, 0, 0, 0};
  v8i z8 = {0, 0, 0, 0, 0, 0, 0, 0};
  __builtin_amdgcn_tensor_load_to_lds(g0, g1, z4, z4, z8, 0);
}

// ---------------- fp32 -> bf16 conversion (vectorized) ---------------------

__global__ __launch_bounds__(256) void f32_to_bf16_kernel(
    const float4* __restrict__ in, v4bf* __restrict__ out, int n4) {
  int i = blockIdx.x * blockDim.x + threadIdx.x;
  int stride = gridDim.x * blockDim.x;
  for (; i < n4; i += stride) {
    float4 x = in[i];
    v4bf y;
    y[0] = (__bf16)x.x; y[1] = (__bf16)x.y;
    y[2] = (__bf16)x.z; y[3] = (__bf16)x.w;
    out[i] = y;
  }
}

// ---------------- GEMM: C[M,N] = A[M,K] @ W[N,K]^T + bias ------------------
// One wave computes a 32x64 tile (2 m-subtiles x 4 n-subtiles, 8 WMMAs per
// 32-K step); fragments double-buffered so loads overlap the WMMA pipe.

__global__ __launch_bounds__(256) void gemm_wmma_bf16(
    const __bf16* __restrict__ A, const __bf16* __restrict__ W,
    const float* __restrict__ bias, float* __restrict__ outF,
    __bf16* __restrict__ outB, int M, int N, int K) {
  int lane = threadIdx.x & 31;
  int wid  = threadIdx.x >> 5;
  int t    = blockIdx.x * 8 + wid;
  int ntn  = N >> 6;
  int mt   = t / ntn;
  int nt   = t - mt * ntn;
  int m0   = mt << 5;                 // 32 rows
  int n0   = nt << 6;                 // 64 cols

  v8f acc[2][4];
#pragma unroll
  for (int s = 0; s < 2; ++s)
#pragma unroll
    for (int c = 0; c < 4; ++c)
#pragma unroll
      for (int r = 0; r < 8; ++r) acc[s][c][r] = 0.0f;

  const __bf16* Ap0 = A + (size_t)m0 * K;
  const __bf16* Ap1 = Ap0 + (size_t)16 * K;

  v16bf a0 = load_a16(Ap0, K);
  v16bf a1 = load_a16(Ap1, K);
  v16bf b[4];
#pragma unroll
  for (int c = 0; c < 4; ++c)
    b[c] = load_b_rm(W + (size_t)(n0 + 16 * c) * K, K);

  for (int k0 = 0; k0 < K - 32; k0 += 32) {
    v16bf a0n = load_a16(Ap0 + k0 + 32, K);
    v16bf a1n = load_a16(Ap1 + k0 + 32, K);
    v16bf bn[4];
#pragma unroll
    for (int c = 0; c < 4; ++c)
      bn[c] = load_b_rm(W + (size_t)(n0 + 16 * c) * K + k0 + 32, K);
#pragma unroll
    for (int c = 0; c < 4; ++c) {
      acc[0][c] = __builtin_amdgcn_wmma_f32_16x16x32_bf16(
          false, a0, false, b[c], (short)0, acc[0][c], false, false);
      acc[1][c] = __builtin_amdgcn_wmma_f32_16x16x32_bf16(
          false, a1, false, b[c], (short)0, acc[1][c], false, false);
    }
    a0 = a0n; a1 = a1n;
#pragma unroll
    for (int c = 0; c < 4; ++c) b[c] = bn[c];
  }
#pragma unroll
  for (int c = 0; c < 4; ++c) {
    acc[0][c] = __builtin_amdgcn_wmma_f32_16x16x32_bf16(
        false, a0, false, b[c], (short)0, acc[0][c], false, false);
    acc[1][c] = __builtin_amdgcn_wmma_f32_16x16x32_bf16(
        false, a1, false, b[c], (short)0, acc[1][c], false, false);
  }

  int n  = lane & 15;
  int mh = (lane >> 4) << 3;
#pragma unroll
  for (int s = 0; s < 2; ++s)
#pragma unroll
    for (int c = 0; c < 4; ++c)
#pragma unroll
      for (int r = 0; r < 8; ++r) {
        int row = m0 + 16 * s + mh + r;
        int col = n0 + 16 * c + n;
        float v = acc[s][c][r] + bias[col];
        if (outF) outF[(size_t)row * N + col] = v;
        else      outB[(size_t)row * N + col] = (__bf16)v;
      }
}

// ---------------- flash attention -----------------------------------------
// One block = one (b, h, 128-query slab); 8 waves, one 16-query tile each.
// Wave 0 drives the TDM: K/V 32-key chunks (4 KB each) DMA'd into
// double-buffered LDS, overlapped with the WMMA/softmax of the previous
// chunk; completion via s_wait_tensorcnt, visibility via block barrier.

__global__ __launch_bounds__(256) void attn_wmma(
    const __bf16* __restrict__ proj, __bf16* __restrict__ attn) {
  __shared__ __bf16 kbuf[2][32][HDIM];   // 2 x 4 KB
  __shared__ __bf16 vbuf[2][32][HDIM];   // 2 x 4 KB
  __shared__ __bf16 pbuf[8][16][32];     // per-wave P staging (8 KB)

  int lane = threadIdx.x & 31;
  int wid  = threadIdx.x >> 5;
  int head = blockIdx.x >> 3;            // 0..127
  int qblk = blockIdx.x & 7;             // 8 blocks of 8 q-tiles per head
  int b    = head >> 4;
  int h    = head & 15;
  int q0   = (qblk * 8 + wid) << 4;      // this wave's 16 query rows

  size_t base = (size_t)b * SEQ * EMB3;
  const __bf16* qp = proj + base + (size_t)h * HDIM;   // Q rows, ld = EMB3
  const __bf16* kp = qp + EMB;                          // K rows
  const __bf16* vp = qp + 2 * EMB;                      // V rows

  v16bf qa0 = load_a16(qp + (size_t)q0 * EMB3, EMB3);
  v16bf qa1 = load_a16(qp + (size_t)q0 * EMB3 + 32, EMB3);

  v8f o[4];
  float m_r[8], l_r[8];
#pragma unroll
  for (int c = 0; c < 4; ++c)
#pragma unroll
    for (int r = 0; r < 8; ++r) o[c][r] = 0.0f;
#pragma unroll
  for (int r = 0; r < 8; ++r) { m_r[r] = -1e30f; l_r[r] = 0.0f; }

  v8f zacc;
#pragma unroll
  for (int r = 0; r < 8; ++r) zacc[r] = 0.0f;

  int mh = (lane >> 4) << 3;
  int nn = lane & 15;

  const int NCH = SEQ / 32;              // 32 chunks of 32 keys

  // prologue: DMA chunk 0 into buffer 0
  if (wid == 0) {
    tdm_load_2d((unsigned)(size_t)&kbuf[0][0][0], kp, HDIM, 32, EMB3);
    tdm_load_2d((unsigned)(size_t)&vbuf[0][0][0], vp, HDIM, 32, EMB3);
  }

  for (int jc = 0; jc < NCH; ++jc) {
    int buf = jc & 1;
    if (wid == 0) {
      if (jc + 1 < NCH) {
        const __bf16* kn = kp + (size_t)(jc + 1) * 32 * EMB3;
        const __bf16* vn = vp + (size_t)(jc + 1) * 32 * EMB3;
        tdm_load_2d((unsigned)(size_t)&kbuf[buf ^ 1][0][0], kn, HDIM, 32, EMB3);
        tdm_load_2d((unsigned)(size_t)&vbuf[buf ^ 1][0][0], vn, HDIM, 32, EMB3);
        __builtin_amdgcn_s_wait_tensorcnt(2);   // chunk jc done, jc+1 in flight
      } else {
        __builtin_amdgcn_s_wait_tensorcnt(0);
      }
    }
    __syncthreads();                           // chunk jc visible to all waves

    const __bf16* kc = &kbuf[buf][0][0];       // 32 keys x 64, ld = 64
    const __bf16* vc = &vbuf[buf][0][0];

    // scores: 16 queries x 32 keys (2 tiles x 2 K-splits)
    v16bf kb00 = load_b_rm(kc, HDIM);
    v16bf kb01 = load_b_rm(kc + 32, HDIM);
    v8f s0 = __builtin_amdgcn_wmma_f32_16x16x32_bf16(false, qa0, false, kb00,
                                                     (short)0, zacc, false, false);
    s0 = __builtin_amdgcn_wmma_f32_16x16x32_bf16(false, qa1, false, kb01,
                                                 (short)0, s0, false, false);
    v16bf kb10 = load_b_rm(kc + 16 * HDIM, HDIM);
    v16bf kb11 = load_b_rm(kc + 16 * HDIM + 32, HDIM);
    v8f s1 = __builtin_amdgcn_wmma_f32_16x16x32_bf16(false, qa0, false, kb10,
                                                     (short)0, zacc, false, false);
    s1 = __builtin_amdgcn_wmma_f32_16x16x32_bf16(false, qa1, false, kb11,
                                                 (short)0, s1, false, false);

    // online softmax (row M spans 16 lanes of a half-wave, one row per VGPR)
#pragma unroll
    for (int r = 0; r < 8; ++r) {
      float a0 = s0[r] * 0.125f;               // 1/sqrt(HDIM)
      float a1 = s1[r] * 0.125f;
      float mx = fmaxf(a0, a1);
#pragma unroll
      for (int off = 8; off; off >>= 1) mx = fmaxf(mx, __shfl_xor(mx, off, 32));
      float mnew  = fmaxf(m_r[r], mx);
      float alpha = __expf(m_r[r] - mnew);
      float p0 = __expf(a0 - mnew);
      float p1 = __expf(a1 - mnew);
      float rs = p0 + p1;
#pragma unroll
      for (int off = 8; off; off >>= 1) rs += __shfl_xor(rs, off, 32);
      l_r[r] = l_r[r] * alpha + rs;
      m_r[r] = mnew;
#pragma unroll
      for (int c = 0; c < 4; ++c) o[c][r] *= alpha;
      pbuf[wid][mh + r][nn]      = (__bf16)p0;
      pbuf[wid][mh + r][16 + nn] = (__bf16)p1;
    }
    __syncthreads();                           // P staged (C-layout -> A-layout)

    v16bf pa = load_a16(&pbuf[wid][0][0], 32);
#pragma unroll
    for (int c = 0; c < 4; ++c) {
      v16bf vb = load_b_cm(vc + c * 16, HDIM);
      o[c] = __builtin_amdgcn_wmma_f32_16x16x32_bf16(false, pa, false, vb,
                                                     (short)0, o[c], false, false);
    }
    __syncthreads();                           // buf free for next DMA reuse
  }

  // normalize and store (heads concatenated along feature dim)
#pragma unroll
  for (int r = 0; r < 8; ++r) {
    float inv = 1.0f / l_r[r];
    int row = q0 + mh + r;
    size_t ob = ((size_t)b * SEQ + row) * EMB + h * HDIM + nn;
#pragma unroll
    for (int c = 0; c < 4; ++c) attn[ob + 16 * c] = (__bf16)(o[c][r] * inv);
  }
}

// ---------------- launch ----------------------------------------------------

extern "C" void kernel_launch(void* const* d_in, const int* in_sizes, int n_in,
                              void* d_out, int out_size, void* d_ws, size_t ws_size,
                              hipStream_t stream) {
  const float* qkv   = (const float*)d_in[0];  // [8,1024,1024]
  const float* w_in  = (const float*)d_in[1];  // [3072,1024]
  const float* b_in  = (const float*)d_in[2];  // [3072]
  const float* w_out = (const float*)d_in[3];  // [1024,1024]
  const float* b_out = (const float*)d_in[4];  // [1024]
  float* out = (float*)d_out;                  // [8,1024,1024]

  const int M = BATCH * SEQ;                   // 8192
  char* ws = (char*)d_ws;
  __bf16* Xb    = (__bf16*)(ws);                                // 16 MiB
  __bf16* Winb  = (__bf16*)(ws + (size_t)16 * 1024 * 1024);     //  6 MiB
  __bf16* Woutb = (__bf16*)(ws + (size_t)22 * 1024 * 1024);     //  2 MiB
  __bf16* projb = (__bf16*)(ws + (size_t)24 * 1024 * 1024);     // 48 MiB
  __bf16* attnb = (__bf16*)(ws + (size_t)72 * 1024 * 1024);     // 16 MiB (total 88 MiB)

  // 1. fp32 -> bf16
  f32_to_bf16_kernel<<<2048, 256, 0, stream>>>((const float4*)qkv,   (v4bf*)Xb,    (M * EMB) / 4);
  f32_to_bf16_kernel<<<1024, 256, 0, stream>>>((const float4*)w_in,  (v4bf*)Winb,  (EMB3 * EMB) / 4);
  f32_to_bf16_kernel<<<512,  256, 0, stream>>>((const float4*)w_out, (v4bf*)Woutb, (EMB * EMB) / 4);

  // 2. in_proj: proj[8192,3072] = X @ W_in^T + b_in (bf16 out)
  //    (8192/32)*(3072/64) = 12288 wave-tiles -> 1536 blocks of 8 waves
  gemm_wmma_bf16<<<1536, 256, 0, stream>>>(Xb, Winb, b_in, nullptr, projb,
                                           M, EMB3, EMB);

  // 3. flash attention: 128 heads x 8 query-slabs = 1024 blocks
  attn_wmma<<<1024, 256, 0, stream>>>(projb, attnb);

  // 4. out_proj: out[8192,1024] = attn @ W_out^T + b_out (fp32 out)
  //    (8192/32)*(1024/64) = 4096 wave-tiles -> 512 blocks
  gemm_wmma_bf16<<<512, 256, 0, stream>>>(attnb, Woutb, b_out, out, nullptr,
                                          M, EMB, EMB);
}
